// att_select_79182017069250
// MI455X (gfx1250) — compile-verified, matrix operand order checked by reading, not run on hardware
//
#include <hip/hip_runtime.h>
#include <hip/hip_bf16.h>

// Shapes from the reference
#define B_  1024
#define N_  60
#define F_  1024

typedef __attribute__((ext_vector_type(16))) __bf16 v16bf;
typedef __attribute__((ext_vector_type(8)))  __bf16 v8bf;
typedef __attribute__((ext_vector_type(8)))  float  v8f;

union Frag16 { v16bf v; v8bf h[2]; };

// Output section offsets (flat f32, return-tuple order)
#define OFF_X    0            // (B,2,2048)
#define OFF_O1   4194304      // (B,2,1024)
#define OFF_O2   6291456
#define OFF_H1   8388608
#define OFF_H2   10485760
#define OFF_ATT1 12582912     // (B,60)
#define OFF_ATT2 12644352
#define OFF_A11  12705792
#define OFF_A22  12767232
#define OFF_A12  12828672
#define OFF_A21  12890112

// ---------------------------------------------------------------------------
// Kernel 1: per-batch logits -> argmax -> gather h rows; write h f32 to out,
// and tanh(h) as bf16 rows into ws (Ht: 4*B rows x 1024, row = b*4 + r,
// r: 0=h1c0 1=h1c1 2=h2c0 3=h2c1)
// ---------------------------------------------------------------------------
__global__ __launch_bounds__(256) void k_select(const float* __restrict__ x1,
                                                const float* __restrict__ x2,
                                                const float* __restrict__ fc_w,
                                                const float* __restrict__ fc_b,
                                                float* __restrict__ h1_out,
                                                float* __restrict__ h2_out,
                                                __bf16* __restrict__ Ht)
{
    const int b    = blockIdx.x;
    const int tid  = threadIdx.x;
    const int wave = tid >> 5;
    const int lane = tid & 31;

    __shared__ float lg[2][N_][2];
    __shared__ int   sidx[4];

    for (int src = 0; src < 2; ++src) {
        const float* xb = (src ? x2 : x1) + (size_t)b * N_ * F_;
        for (int p = wave; p < N_ * 2; p += 8) {
            const int n = p >> 1, c = p & 1;
            const float* row = xb + (size_t)n * F_;
            const float* w   = fc_w + c * F_;
            float s = 0.f;
            for (int f = lane; f < F_; f += 32) s += row[f] * w[f];
            for (int m = 16; m >= 1; m >>= 1) s += __shfl_xor(s, m, 32);
            if (lane == 0) lg[src][n][c] = s + fc_b[c];
        }
    }
    __syncthreads();

    if (tid < 4) {                       // argmax over N (first-max tiebreak)
        const int src = tid >> 1, c = tid & 1;
        float best = lg[src][0][c];
        int   bi   = 0;
        for (int n = 1; n < N_; ++n) {
            const float v = lg[src][n][c];
            if (v > best) { best = v; bi = n; }
        }
        sidx[tid] = bi;
    }
    __syncthreads();

    for (int r = 0; r < 4; ++r) {
        const int src = r >> 1, c = r & 1;
        const int n   = sidx[r];
        const float* row  = (src ? x2 : x1) + ((size_t)b * N_ + n) * F_;
        float*       hout = (src ? h2_out : h1_out) + ((size_t)b * 2 + c) * F_;
        __bf16*      ht   = Ht + ((size_t)b * 4 + r) * F_;
        for (int f = tid; f < F_; f += 256) {
            const float v = row[f];
            hout[f] = v;
            ht[f]   = (__bf16)tanhf(v);
        }
    }
}

// ---------------------------------------------------------------------------
// Kernel 2a: q_w f32 -> bf16
// ---------------------------------------------------------------------------
__global__ __launch_bounds__(256) void k_cvt_w(const float* __restrict__ qw,
                                               __bf16* __restrict__ Wb)
{
    const int stride = gridDim.x * 256;
    for (int i = blockIdx.x * 256 + threadIdx.x; i < F_ * F_; i += stride)
        Wb[i] = (__bf16)qw[i];
}

// ---------------------------------------------------------------------------
// Kernel 2: Qm(4096x1024) = Ht(4096x1024,bf16) @ q_w^T(bf16) + q_b, f32 acc.
// One 16x16 tile per wave, K-loop of 32 with v_wmma_f32_16x16x32_bf16.
// Fragment addressing follows the ISA 16-bit A layout: lane L holds row
// (L&15), K-halves {kb..kb+7} and {kb+16..kb+23} with kb = (L>>4)*8.
// B is loaded from q_w rows (K-contiguous == B^T row-major), same pattern.
// ---------------------------------------------------------------------------
__global__ __launch_bounds__(256) void k_qgemm(const __bf16* __restrict__ Ht,
                                               const __bf16* __restrict__ Wb,
                                               const float* __restrict__ qb,
                                               float* __restrict__ Qm)
{
    const int wave = threadIdx.x >> 5;
    const int lane = threadIdx.x & 31;
    const int tile = blockIdx.x * 8 + wave;   // 16384 tiles total
    const int tn   = tile & 63;               // 64 N-tiles (1024/16)
    const int tm   = tile >> 6;               // 256 M-tiles (4096/16)

    const int row = lane & 15;
    const int kb  = (lane >> 4) * 8;
    const __bf16* aptr = Ht + ((size_t)(tm * 16 + row)) * F_ + kb;
    const __bf16* bptr = Wb + ((size_t)(tn * 16 + row)) * F_ + kb;

    v8f acc = {};
    for (int k = 0; k < F_; k += 32) {
        Frag16 a, bfr;
        a.h[0]   = *(const v8bf*)(aptr + k);
        a.h[1]   = *(const v8bf*)(aptr + k + 16);
        bfr.h[0] = *(const v8bf*)(bptr + k);
        bfr.h[1] = *(const v8bf*)(bptr + k + 16);
        acc = __builtin_amdgcn_wmma_f32_16x16x32_bf16(
            false, a.v, false, bfr.v, (short)0, acc, false, false);
    }

    // D layout: VGPR g -> M = g (lanes 0-15) or 8+g (lanes 16-31), N = lane&15
    const int j     = tn * 16 + (lane & 15);
    const float bias = qb[j];
    const int mbase = tm * 16 + ((lane >> 4) ? 8 : 0);
    for (int g = 0; g < 8; ++g)
        Qm[(size_t)(mbase + g) * F_ + j] = acc[g] + bias;
}

// ---------------------------------------------------------------------------
// Kernel 3: per-batch fused scores -> softmax -> weighted sum + all outputs
// ---------------------------------------------------------------------------
__global__ __launch_bounds__(256) void k_attend(const float* __restrict__ x1,
                                                const float* __restrict__ x2,
                                                const float* __restrict__ Qm,
                                                float* __restrict__ out)
{
    const int b    = blockIdx.x;
    const int tid  = threadIdx.x;
    const int wave = tid >> 5;
    const int lane = tid & 31;

    __shared__ float sA[2 * N_][4];          // scores -> softmax(A) in place
    const float* qmb = Qm + (size_t)b * 4 * F_;

    // scores: S[n][cc] = (1/32) * dot(L[n], qm[cc]); cc: 0,1 = qm1; 2,3 = qm2
    for (int p = wave; p < 2 * N_ * 4; p += 8) {
        const int n = p >> 2, cc = p & 3;
        const float* Lrow = (n < N_) ? x1 + ((size_t)b * N_ + n) * F_
                                     : x2 + ((size_t)b * N_ + (n - N_)) * F_;
        const float* q = qmb + cc * F_;
        float s = 0.f;
        for (int f = lane; f < F_; f += 32) s += Lrow[f] * q[f];
        for (int m = 16; m >= 1; m >>= 1) s += __shfl_xor(s, m, 32);
        if (lane == 0) sA[n][cc] = s * 0.03125f;
    }
    __syncthreads();

    if (tid < 4) {                            // softmax over the 120 rows
        const int cc = tid;
        float mx = sA[0][cc];
        for (int n = 1; n < 2 * N_; ++n) mx = fmaxf(mx, sA[n][cc]);
        float sum = 0.f;
        for (int n = 0; n < 2 * N_; ++n) {
            const float e = __expf(sA[n][cc] - mx);
            sA[n][cc] = e;
            sum += e;
        }
        const float inv = 1.f / sum;
        for (int n = 0; n < 2 * N_; ++n) sA[n][cc] *= inv;
    }
    __syncthreads();

    float* out_x  = out + OFF_X;
    float* out_o1 = out + OFF_O1;
    float* out_o2 = out + OFF_O2;

    for (int f = tid; f < F_; f += 256) {
        float a0 = 0.f, a1 = 0.f, a2 = 0.f, a3 = 0.f;
        for (int n = 0; n < N_; ++n) {
            const float v1 = x1[((size_t)b * N_ + n) * F_ + f];
            const float v2 = x2[((size_t)b * N_ + n) * F_ + f];
            a0 += sA[n][0] * v1 + sA[N_ + n][0] * v2;
            a1 += sA[n][1] * v1 + sA[N_ + n][1] * v2;
            a2 += sA[n][2] * v1 + sA[N_ + n][2] * v2;
            a3 += sA[n][3] * v1 + sA[N_ + n][3] * v2;
        }
        const float o1c0 = 0.9f * a0 + 0.1f * qmb[0 * F_ + f];
        const float o1c1 = 0.9f * a1 + 0.1f * qmb[1 * F_ + f];
        const float o2c0 = 0.9f * a2 + 0.1f * qmb[2 * F_ + f];
        const float o2c1 = 0.9f * a3 + 0.1f * qmb[3 * F_ + f];

        out_o1[((size_t)b * 2 + 0) * F_ + f] = o1c0;
        out_o1[((size_t)b * 2 + 1) * F_ + f] = o1c1;
        out_o2[((size_t)b * 2 + 0) * F_ + f] = o2c0;
        out_o2[((size_t)b * 2 + 1) * F_ + f] = o2c1;
        out_x[((size_t)b * 2 + 0) * 2048 + f]      = o1c0;
        out_x[((size_t)b * 2 + 0) * 2048 + F_ + f] = o2c0;
        out_x[((size_t)b * 2 + 1) * 2048 + f]      = o1c1;
        out_x[((size_t)b * 2 + 1) * 2048 + F_ + f] = o2c1;
    }

    if (tid < N_) {
        const int n = tid;
        out[OFF_ATT1 + (size_t)b * N_ + n] = sA[n][0];
        out[OFF_ATT2 + (size_t)b * N_ + n] = sA[N_ + n][2];
        out[OFF_A11  + (size_t)b * N_ + n] = sA[n][1];
        out[OFF_A22  + (size_t)b * N_ + n] = sA[N_ + n][3];
        out[OFF_A12  + (size_t)b * N_ + n] = sA[N_ + n][1];
        out[OFF_A21  + (size_t)b * N_ + n] = sA[n][3];
    }
}

// ---------------------------------------------------------------------------
extern "C" void kernel_launch(void* const* d_in, const int* in_sizes, int n_in,
                              void* d_out, int out_size, void* d_ws, size_t ws_size,
                              hipStream_t stream)
{
    const float* x1   = (const float*)d_in[0];
    const float* x2   = (const float*)d_in[1];
    const float* fc_w = (const float*)d_in[2];
    const float* fc_b = (const float*)d_in[3];
    const float* q_w  = (const float*)d_in[4];
    const float* q_b  = (const float*)d_in[5];
    float* out = (float*)d_out;

    char* ws = (char*)d_ws;
    __bf16* Ht = (__bf16*)ws;                        //  8 MB: tanh(h) bf16
    __bf16* Wb = (__bf16*)(ws + ((size_t)8  << 20)); //  2 MB: q_w bf16
    float*  Qm = (float*) (ws + ((size_t)10 << 20)); // 16 MB: qm f32

    k_select<<<B_, 256, 0, stream>>>(x1, x2, fc_w, fc_b,
                                     out + OFF_H1, out + OFF_H2, Ht);
    k_cvt_w<<<1024, 256, 0, stream>>>(q_w, Wb);
    k_qgemm<<<2048, 256, 0, stream>>>(Ht, Wb, q_b, Qm);   // 16384 tiles / 8 waves
    k_attend<<<B_, 256, 0, stream>>>(x1, x2, Qm, out);
}
